// LSTM_Autoencoder_72808285601822
// MI455X (gfx1250) — matrix operand order repaired and checked by reading, not compiled, not run on hardware
//
#include <hip/hip_runtime.h>

// ---------------------------------------------------------------------------
// 6-layer LSTM autoencoder, fused recurrent scan with bf16 WMMA (gfx1250).
// B=128, T=512; layers (Din,H): (64,256)(256,256)(256,128)(128,128)(128,256)(256,64)
// Weight slices staged to LDS via TDM (tensor_load_to_lds) with LDS row padding.
// ---------------------------------------------------------------------------

#define B_ 128
#define T_ 512

typedef __attribute__((ext_vector_type(16))) __bf16   v16bf;
typedef __attribute__((ext_vector_type(8)))  float    v8f;
typedef __attribute__((ext_vector_type(4)))  unsigned u32x4;
typedef __attribute__((ext_vector_type(8)))  unsigned u32x8;

union ABfrag { v16bf v; u32x4 q[2]; };

__device__ __forceinline__ float sigmoidf_(float x) {
  return 1.0f / (1.0f + __expf(-x));
}
__device__ __forceinline__ float actf_(float x, int act) {
  return act == 0 ? fmaxf(x, 0.0f) : sigmoidf_(x);
}

__global__ void k_init_barrier(int* cnt, unsigned* gen) {
  if (threadIdx.x == 0) { *cnt = 0; *gen = 0u; }
}

__global__ void k_cvt_bf16(const float* __restrict__ src, __bf16* __restrict__ dst, int n) {
  int i = blockIdx.x * blockDim.x + threadIdx.x;
  if (i < n) dst[i] = (__bf16)src[i];
}

// W: [K][N] row-major (f32) -> WT: [N][K] (bf16), K-contiguous rows for WMMA B-frags
__global__ void k_cvt_transpose(const float* __restrict__ W, __bf16* __restrict__ WT,
                                int K, int N) {
  int i = blockIdx.x * blockDim.x + threadIdx.x;
  if (i >= K * N) return;
  int n = i / K, k = i - n * K;
  WT[i] = (__bf16)W[(size_t)k * N + n];
}

// ---------------------------------------------------------------------------
// TDM 2D tile load: 'rows' rows of 'tile_w' bf16 elements, global row stride
// 'stride_elems', into LDS at lds_addr with pad_amount DWORDs of padding
// inserted every pad_interval (both as D# codes) -- packs Wx|Wh rows into one
// concatenated Ktot-stride LDS row.  D# layout per CDNA5 ISA.
// ---------------------------------------------------------------------------
__device__ __forceinline__ void tdm_load_2d(
    unsigned lds_addr, unsigned long long gaddr,
    unsigned tile_w, unsigned rows, unsigned stride_elems,
    unsigned pad_iv_code, unsigned pad_amt_code)
{
  u32x4 g0;
  g0.x = 1u;                                              // count=1 (valid D#)
  g0.y = lds_addr;                                        // lds_addr (bytes)
  g0.z = (unsigned)gaddr;                                 // global_addr[31:0]
  g0.w = (unsigned)((gaddr >> 32) & 0x01ffffffull)        // global_addr[56:32]
       | (2u << 30);                                      // type=2 ("image")
  u32x8 g1;
  g1[0] = (1u << 16)                                      // data_size=1 (2 bytes)
        | (1u << 20)                                      // pad_enable
        | (pad_iv_code << 22)                             // pad_interval
        | (pad_amt_code << 25);                           // pad_amount
  g1[1] = (tile_w & 0xffffu) << 16;                       // tensor_dim0[15:0]
  g1[2] = (tile_w >> 16) | ((rows & 0xffffu) << 16);      // td0 hi | tensor_dim1 lo
  g1[3] = (rows >> 16) | ((tile_w & 0xffffu) << 16);      // td1 hi | tile_dim0
  g1[4] = rows & 0xffffu;                                 // tile_dim1 (tile_dim2=0)
  g1[5] = stride_elems;                                   // tensor_dim0_stride[31:0]
  g1[6] = 0u;                                             // stride hi | dim1_stride lo
  g1[7] = 0u;
  asm volatile("tensor_load_to_lds %0, %1" :: "s"(g0), "s"(g1) : "memory");
}

// log2(row_bytes/8) as pad_interval code; dims are 64/128/256 only
__device__ __forceinline__ unsigned pad_iv_code_(int elems) {
  return (elems == 64) ? 4u : (elems == 128) ? 5u : 6u;
}

// One launch per layer. grid = H/16 workgroups, 256 threads (8 waves).
// Wave w of block j handles batch-tile w, output columns [16j,16j+16) x 4 gates.
__global__ void __launch_bounds__(256)
k_lstm_scan(const __bf16* __restrict__ in, int Din,
            const __bf16* __restrict__ wxT, const __bf16* __restrict__ whT,
            const float* __restrict__ bias, int H,
            __bf16* __restrict__ hbuf,      // 2 * 128*256 bf16 (double buffered)
            __bf16* __restrict__ outb, float* __restrict__ outf,
            int act, int* cnt, unsigned* gen)
{
  // 64 output columns (4 gates x 16) x Ktot bf16, compact row stride = Ktot
  __shared__ __bf16 lds_w[64 * 512];

  const int tid   = threadIdx.x;
  const int lane  = tid & 31;
  const int wave  = tid >> 5;
  const int lm    = lane & 15;   // row-in-tile (A) / column (B/C)
  const int grp   = lane >> 4;   // K-half selector
  const int jbase = blockIdx.x * 16;
  const int mbase = wave * 16;
  const int Ktot  = Din + H;
  const int nwg   = gridDim.x;
  const int HB    = 128 * 256;   // hbuf buffer stride (elements)

  // ---- TDM-stage this WG's weight slice into LDS (wave 0 issues, all wait)
  if (wave == 0) {
    unsigned ldsbase = (unsigned)(uintptr_t)(void*)&lds_w[0];
    unsigned iv_x = pad_iv_code_(Din), iv_h = pad_iv_code_(H);
    unsigned pa_x = (unsigned)(H / 2 - 1);    // pad after Wx row: H elems
    unsigned pa_h = (unsigned)(Din / 2 - 1);  // pad after Wh row: Din elems
    #pragma unroll
    for (int gi = 0; gi < 4; gi++) {
      unsigned lrow = ldsbase + (unsigned)(gi * 16) * (unsigned)Ktot * 2u;
      tdm_load_2d(lrow,
                  (unsigned long long)(uintptr_t)(wxT + (size_t)(gi * H + jbase) * Din),
                  (unsigned)Din, 16u, (unsigned)Din, iv_x, pa_x);
      tdm_load_2d(lrow + (unsigned)Din * 2u,
                  (unsigned long long)(uintptr_t)(whT + (size_t)(gi * H + jbase) * H),
                  (unsigned)H, 16u, (unsigned)H, iv_h, pa_h);
    }
    __builtin_amdgcn_s_wait_tensorcnt(0);
  }
  __syncthreads();

  // per-lane bias (column jbase+lm), Keras gate order [i,f,g,o]
  float bv[4];
  #pragma unroll
  for (int gi = 0; gi < 4; gi++) bv[gi] = bias[gi * H + jbase + lm];

  // per-gate LDS row base for this lane's B-column
  const __bf16* wrow0 = lds_w + (0 * 16 + lm) * Ktot;
  const __bf16* wrow1 = lds_w + (1 * 16 + lm) * Ktot;
  const __bf16* wrow2 = lds_w + (2 * 16 + lm) * Ktot;
  const __bf16* wrow3 = lds_w + (3 * 16 + lm) * Ktot;

  const __bf16* xrow0 = in + (size_t)(mbase + lm) * T_ * Din;

  v8f cacc;                       // cell state, f32, lives in registers
  #pragma unroll
  for (int e = 0; e < 8; e++) cacc[e] = 0.0f;

  for (int t = 0; t < T_; t++) {
    v8f accs[4];
    #pragma unroll
    for (int gi = 0; gi < 4; gi++) {
      #pragma unroll
      for (int e = 0; e < 8; e++) accs[gi][e] = bv[gi];
    }

    // one K-step: preload A + all 4 B frags, then 4 back-to-back WMMAs
#define GEMM_K_STEP(AROW, WOFF)                                              \
    {                                                                        \
      ABfrag a_, b0_, b1_, b2_, b3_;                                         \
      a_.q[0]  = *(const u32x4*)((AROW) + kb + grp * 16);                    \
      a_.q[1]  = *(const u32x4*)((AROW) + kb + grp * 16 + 8);                \
      b0_.q[0] = *(const u32x4*)(wrow0 + (WOFF) + kb + grp * 16);            \
      b0_.q[1] = *(const u32x4*)(wrow0 + (WOFF) + kb + grp * 16 + 8);        \
      b1_.q[0] = *(const u32x4*)(wrow1 + (WOFF) + kb + grp * 16);            \
      b1_.q[1] = *(const u32x4*)(wrow1 + (WOFF) + kb + grp * 16 + 8);        \
      b2_.q[0] = *(const u32x4*)(wrow2 + (WOFF) + kb + grp * 16);            \
      b2_.q[1] = *(const u32x4*)(wrow2 + (WOFF) + kb + grp * 16 + 8);        \
      b3_.q[0] = *(const u32x4*)(wrow3 + (WOFF) + kb + grp * 16);            \
      b3_.q[1] = *(const u32x4*)(wrow3 + (WOFF) + kb + grp * 16 + 8);        \
      accs[0] = __builtin_amdgcn_wmma_f32_16x16x32_bf16(                     \
          false, a_.v, false, b0_.v, (short)0, accs[0], false, false);       \
      accs[1] = __builtin_amdgcn_wmma_f32_16x16x32_bf16(                     \
          false, a_.v, false, b1_.v, (short)0, accs[1], false, false);       \
      accs[2] = __builtin_amdgcn_wmma_f32_16x16x32_bf16(                     \
          false, a_.v, false, b2_.v, (short)0, accs[2], false, false);       \
      accs[3] = __builtin_amdgcn_wmma_f32_16x16x32_bf16(                     \
          false, a_.v, false, b3_.v, (short)0, accs[3], false, false);       \
    }

    // ---- x_t @ Wx  (LDS K-offset 0..Din)
    const __bf16* arow = xrow0 + (size_t)t * Din;
    if (t + 1 < T_) __builtin_prefetch(arow + Din, 0, 1);  // next timestep row
    for (int kb = 0; kb < Din; kb += 32) {
      GEMM_K_STEP(arow, 0)
    }

    // ---- h_{t-1} @ Wh  (LDS K-offset Din..Din+H), previous double-buffer half
    if (t > 0) {
      const __bf16* hprev = hbuf + ((t & 1) ^ 1) * HB + (size_t)(mbase + lm) * H;
      for (int kb = 0; kb < H; kb += 32) {
        GEMM_K_STEP(hprev, Din)
      }
    }
#undef GEMM_K_STEP

    // ---- gates + state update + stores (C-layout: M = mbase + grp*8 + r, N = jbase+lm)
    __bf16* hcur = hbuf + (t & 1) * HB;
    const int jc = jbase + lm;
    #pragma unroll
    for (int r = 0; r < 8; r++) {
      float iv = sigmoidf_(accs[0][r]);
      float fv = sigmoidf_(accs[1][r]);
      float gv = actf_(accs[2][r], act);
      float ov = sigmoidf_(accs[3][r]);
      float cv = fv * cacc[r] + iv * gv;
      cacc[r] = cv;
      float hv = ov * actf_(cv, act);
      int Mg = mbase + grp * 8 + r;
      hcur[(size_t)Mg * H + jc] = (__bf16)hv;
      if (outf) outf[((size_t)Mg * T_ + t) * H + jc] = hv;
      else      outb[((size_t)Mg * T_ + t) * H + jc] = (__bf16)hv;
    }

    // ---- grid-wide barrier between timesteps (sense-reversing, atomics in ws)
    __threadfence();
    __syncthreads();
    if (tid == 0) {
      unsigned g = atomicAdd(gen, 0u);
      if (atomicAdd(cnt, 1) == nwg - 1) {
        atomicExch(cnt, 0);
        __threadfence();
        atomicAdd(gen, 1u);
      } else {
        while (atomicAdd(gen, 0u) == g) { __builtin_amdgcn_s_sleep(2); }
      }
    }
    __syncthreads();
  }
}

extern "C" void kernel_launch(void* const* d_in, const int* in_sizes, int n_in,
                              void* d_out, int out_size, void* d_ws, size_t ws_size,
                              hipStream_t stream) {
  (void)in_sizes; (void)n_in; (void)out_size; (void)ws_size;
  static const int DIN[6] = {64, 256, 256, 128, 128, 256};
  static const int HH [6] = {256, 256, 128, 128, 256, 64};

  char* ws = (char*)d_ws;
  size_t off = 0;
  auto alloc = [&](size_t bytes) {
    size_t r = off; off += (bytes + 255) & ~(size_t)255; return r;
  };

  size_t bar_off  = alloc(256);
  size_t hbuf_off = alloc((size_t)2 * 128 * 256 * sizeof(__bf16));
  size_t actA_off = alloc((size_t)128 * 512 * 256 * sizeof(__bf16));  // 32 MB
  size_t actB_off = alloc((size_t)128 * 512 * 256 * sizeof(__bf16));  // 32 MB
  size_t wxt_off[6], wht_off[6];
  for (int l = 0; l < 6; l++) {
    wxt_off[l] = alloc((size_t)4 * HH[l] * DIN[l] * sizeof(__bf16));
    wht_off[l] = alloc((size_t)4 * HH[l] * HH[l] * sizeof(__bf16));
  }

  int*      cnt  = (int*)(ws + bar_off);
  unsigned* gen  = (unsigned*)(ws + bar_off + 64);
  __bf16*   hbuf = (__bf16*)(ws + hbuf_off);
  __bf16*   actA = (__bf16*)(ws + actA_off);
  __bf16*   actB = (__bf16*)(ws + actB_off);

  k_init_barrier<<<1, 64, 0, stream>>>(cnt, gen);

  // x -> bf16 into actA
  {
    int n = 128 * 512 * 64;
    k_cvt_bf16<<<(n + 255) / 256, 256, 0, stream>>>((const float*)d_in[0], actA, n);
  }
  // weights -> transposed bf16 [4H][K]
  for (int l = 0; l < 6; l++) {
    const float* Wx = (const float*)d_in[1 + 3 * l];
    const float* Wh = (const float*)d_in[2 + 3 * l];
    int N = 4 * HH[l];
    int nx = DIN[l] * N;
    k_cvt_transpose<<<(nx + 255) / 256, 256, 0, stream>>>(
        Wx, (__bf16*)(ws + wxt_off[l]), DIN[l], N);
    int nh = HH[l] * N;
    k_cvt_transpose<<<(nh + 255) / 256, 256, 0, stream>>>(
        Wh, (__bf16*)(ws + wht_off[l]), HH[l], N);
  }

  // sequential layers, ping-pong activation buffers
  __bf16* cur = actA;
  __bf16* nxt = actB;
  for (int l = 0; l < 6; l++) {
    const float* bias = (const float*)d_in[3 + 3 * l];
    int grid = HH[l] / 16;
    __bf16* outb = (l == 5) ? nullptr : nxt;
    float*  outf = (l == 5) ? (float*)d_out : nullptr;
    k_lstm_scan<<<grid, 256, 0, stream>>>(
        cur, DIN[l],
        (const __bf16*)(ws + wxt_off[l]), (const __bf16*)(ws + wht_off[l]),
        bias, HH[l], hbuf, outb, outf, (l == 5) ? 1 : 0, cnt, gen);
    __bf16* t = nxt; nxt = cur; cur = t;
  }
}
